// EMA_Quantizer_17995912970285
// MI455X (gfx1250) — matrix operand order repaired and checked by reading, not compiled, not run on hardware
//
#include <hip/hip_runtime.h>
#include <hip/hip_bf16.h>
#include <math.h>

// ---------------- problem constants ----------------
#define NTOK   16384      // 16*32*32 tokens
#define DDIM   256
#define KCODE  16384
#define DECAYF 0.8f
#define ONEMD  (1.0f - 0.8f)
#define BETAF  0.25f
#define EPSF   1e-5f

typedef __attribute__((ext_vector_type(16))) __bf16 bf16x16;
typedef __attribute__((ext_vector_type(8)))  __bf16 bf16x8;
typedef __attribute__((ext_vector_type(8)))  float  f32x8;
typedef __attribute__((address_space(3)))    unsigned short lds_ushort;

union AFrag { bf16x16 v; bf16x8 h[2]; };

__device__ __forceinline__ unsigned short f2bf(float f) {
    unsigned u = __float_as_uint(f);
    unsigned r = u + 0x7FFFu + ((u >> 16) & 1u);   // round-to-nearest-even
    return (unsigned short)(r >> 16);
}

// ---------------- kernel 1: per-call accumulator init ----------------
__global__ __launch_bounds__(256) void initk(const float* __restrict__ cluster_size,
                                             const float* __restrict__ embed_avg,
                                             float* __restrict__ o_cluster,
                                             float* __restrict__ o_embed,
                                             float* __restrict__ counts,
                                             float* __restrict__ scalars) {
    size_t i = (size_t)blockIdx.x * 256 + threadIdx.x;
    if (i < (size_t)KCODE * DDIM) o_embed[i] = DECAYF * embed_avg[i];
    if (i < KCODE) { o_cluster[i] = DECAYF * cluster_size[i]; counts[i] = 0.0f; }
    if (i < 4) scalars[i] = 0.0f;
}

// ---------------- kernel 2: weight f32 -> bf16 + |w|^2 ----------------
__global__ __launch_bounds__(256) void wprep(const float* __restrict__ w,
                                             unsigned short* __restrict__ wbf,
                                             float* __restrict__ wnorm) {
    int gid  = blockIdx.x * 256 + threadIdx.x;
    int row  = gid >> 5;                 // one wave per codebook row
    int lane = threadIdx.x & 31;
    if (row >= KCODE) return;
    const float* rp = w + (size_t)row * DDIM + lane * 8;
    float4 v0 = *(const float4*)(rp);
    float4 v1 = *(const float4*)(rp + 4);
    float s = v0.x*v0.x + v0.y*v0.y + v0.z*v0.z + v0.w*v0.w
            + v1.x*v1.x + v1.y*v1.y + v1.z*v1.z + v1.w*v1.w;
    unsigned short h[8] = { f2bf(v0.x), f2bf(v0.y), f2bf(v0.z), f2bf(v0.w),
                            f2bf(v1.x), f2bf(v1.y), f2bf(v1.z), f2bf(v1.w) };
    uint4 o;
    o.x = (unsigned)h[0] | ((unsigned)h[1] << 16);
    o.y = (unsigned)h[2] | ((unsigned)h[3] << 16);
    o.z = (unsigned)h[4] | ((unsigned)h[5] << 16);
    o.w = (unsigned)h[6] | ((unsigned)h[7] << 16);
    *(uint4*)(wbf + (size_t)row * DDIM + lane * 8) = o;
    #pragma unroll
    for (int m = 16; m >= 1; m >>= 1) s += __shfl_xor(s, m, 32);
    if (lane == 0) wnorm[row] = s;
}

// ---------------- kernel 3: (b,c,h,w) -> [N,D] bf16 transpose ----------------
__global__ __launch_bounds__(256) void ztrans(const float* __restrict__ z,
                                              unsigned short* __restrict__ zf) {
    __shared__ float tile[32][33];
    int b  = blockIdx.z;
    int m0 = blockIdx.x * 32;     // hw index within 1024
    int c0 = blockIdx.y * 32;     // channel
    const float* zb = z + (size_t)b * (DDIM * 1024);
    #pragma unroll
    for (int j = 0; j < 4; ++j) {
        int cl = threadIdx.y * 4 + j;
        tile[cl][threadIdx.x] = zb[(size_t)(c0 + cl) * 1024 + m0 + threadIdx.x];
    }
    __syncthreads();
    #pragma unroll
    for (int j = 0; j < 4; ++j) {
        int ml = threadIdx.y * 4 + j;
        float v = tile[threadIdx.x][ml];
        zf[(size_t)(b * 1024 + m0 + ml) * DDIM + c0 + threadIdx.x] = f2bf(v);
    }
}

// ---------------- kernel 4: bf16 WMMA GEMM + argmin ----------------
#define GBLK  128                 // threads per block (4 waves)
#define MTILE 128                 // tokens per workgroup (4 waves x 32)
#define LDS_HSTR 264              // halves per staged row: 256 + 8 pad (528 B)
#define CHUNK 32                  // codes staged per LDS buffer (2 x 16 subtiles)

// async DMA: 16 B global -> LDS, no VGPR data path (ASYNCcnt tracked)
__device__ __forceinline__ void async_cp16(unsigned short* dst, const unsigned short* src) {
    lds_ushort* d3 = (lds_ushort*)dst;   // addrspace(3) ptr == LDS byte offset
    asm volatile("global_load_async_to_lds_b128 %0, %1, off"
                 :: "v"(d3), "v"(src) : "memory");
}

__device__ __forceinline__ void stage_async(const unsigned short* __restrict__ wbf,
                                            unsigned short* dstbase, int chunk, int tid) {
    const int row = tid >> 2;            // 0..31 codes
    const int d0  = (tid & 3) * 64;      // half offset within row (128 B per thread)
    const unsigned short* src = wbf + (size_t)(chunk * CHUNK + row) * DDIM + d0;
    unsigned short* dst = dstbase + row * LDS_HSTR + d0;
    #pragma unroll
    for (int j = 0; j < 8; ++j) async_cp16(dst + j * 8, src + j * 8);
}

__global__ __launch_bounds__(GBLK)
void gemm_argmin(const unsigned short* __restrict__ zf,
                 const unsigned short* __restrict__ wbf,
                 const float* __restrict__ wnorm,
                 unsigned* __restrict__ token) {
    __shared__ __align__(16) unsigned short lds[2][CHUNK * LDS_HSTR];

    const int tid  = threadIdx.x;
    const int lane = tid & 31;
    const int wave = tid >> 5;
    const int col  = lane & 15;       // B column / A row / C column
    const int hi   = lane >> 4;       // K-half selector
    const int rowbase = blockIdx.x * MTILE + wave * 32;   // 32 tokens per wave

    // Two A banks (rows rowbase..+15 and rowbase+16..+31), all of D=256, in VGPRs.
    // ISA 16-bit A 16x32 layout: el0..7 -> K = d0 + hi*8 + e ; el8..15 -> K = d0+16+hi*8+(e-8)
    AFrag a0[8], a1[8];
    {
        const unsigned short* zr0 = zf + (size_t)(rowbase + col) * DDIM;
        const unsigned short* zr1 = zf + (size_t)(rowbase + 16 + col) * DDIM;
        #pragma unroll
        for (int c = 0; c < 8; ++c) {
            a0[c].h[0] = *(const bf16x8*)(zr0 + c * 32 + hi * 8);
            a0[c].h[1] = *(const bf16x8*)(zr0 + c * 32 + 16 + hi * 8);
            a1[c].h[0] = *(const bf16x8*)(zr1 + c * 32 + hi * 8);
            a1[c].h[1] = *(const bf16x8*)(zr1 + c * 32 + 16 + hi * 8);
        }
    }

    float    best0[8], best1[8];
    unsigned btok0[8], btok1[8];
    #pragma unroll
    for (int r = 0; r < 8; ++r) {
        best0[r] = 3.4e38f; btok0[r] = 0u;
        best1[r] = 3.4e38f; btok1[r] = 0u;
    }

    // kick off DMA of chunk 0 into buffer 0
    stage_async(wbf, &lds[0][0], 0, tid);

    const int T = KCODE / CHUNK;         // 512 chunks
    for (int t = 0; t < T; ++t) {
        asm volatile("s_wait_asynccnt 0x0" ::: "memory");   // my DMA for lds[t&1] done
        __syncthreads();                                    // everyone's DMA done; prior reads of other buf done
        if (t + 1 < T) stage_async(wbf, &lds[(t + 1) & 1][0], t + 1, tid);
        #pragma unroll
        for (int sub = 0; sub < 2; ++sub) {
            // B fragment base: column = col, K = hi*8..hi*8+7 and 16+hi*8..16+hi*8+7 within each 32-chunk
            const unsigned short* bbase =
                &lds[t & 1][(sub * 16 + col) * LDS_HSTR + hi * 8];
            f32x8 acc0 = {}, acc1 = {};
            #pragma unroll
            for (int c = 0; c < 8; ++c) {
                AFrag b;                                   // one B feeds two WMMAs
                b.h[0] = *(const bf16x8*)(bbase + c * 32);
                b.h[1] = *(const bf16x8*)(bbase + c * 32 + 16);
                acc0 = __builtin_amdgcn_wmma_f32_16x16x32_bf16(
                           false, a0[c].v, false, b.v, (short)0, acc0, false, false);
                acc1 = __builtin_amdgcn_wmma_f32_16x16x32_bf16(
                           false, a1[c].v, false, b.v, (short)0, acc1, false, false);
            }
            unsigned kbase = (unsigned)(t * CHUNK + sub * 16);
            float wn = wnorm[kbase + col];
            unsigned tk = kbase + (unsigned)col;
            #pragma unroll
            for (int r = 0; r < 8; ++r) {
                // strict < keeps lowest code index per lane (codes scanned ascending)
                float s0 = fmaf(-2.0f, acc0[r], wn);   // |z|^2 dropped (row-constant)
                if (s0 < best0[r]) { best0[r] = s0; btok0[r] = tk; }
                float s1 = fmaf(-2.0f, acc1[r], wn);
                if (s1 < best1[r]) { best1[r] = s1; btok1[r] = tk; }
            }
        }
    }

    // reduce argmin across the 16 columns held by each half-wave
    // (index tie-break only needed here, once)
    #pragma unroll
    for (int r = 0; r < 8; ++r) {
        float s = best0[r]; unsigned tk = btok0[r];
        #pragma unroll
        for (int m = 8; m >= 1; m >>= 1) {
            float    s2 = __shfl_xor(s, m, 32);
            unsigned t2 = (unsigned)__shfl_xor((int)tk, m, 32);
            if (s2 < s || (s2 == s && t2 < tk)) { s = s2; tk = t2; }
        }
        if (col == 0) token[rowbase + hi * 8 + r] = tk;        // lanes 0 and 16
    }
    #pragma unroll
    for (int r = 0; r < 8; ++r) {
        float s = best1[r]; unsigned tk = btok1[r];
        #pragma unroll
        for (int m = 8; m >= 1; m >>= 1) {
            float    s2 = __shfl_xor(s, m, 32);
            unsigned t2 = (unsigned)__shfl_xor((int)tk, m, 32);
            if (s2 < s || (s2 == s && t2 < tk)) { s = s2; tk = t2; }
        }
        if (col == 0) token[rowbase + 16 + hi * 8 + r] = tk;   // lanes 0 and 16
    }
}

// ---------------- kernel 5: gather z_q, residual, EMA scatter ----------------
__global__ __launch_bounds__(256) void scatter(const float* __restrict__ z,
                                               const float* __restrict__ weight,
                                               const unsigned* __restrict__ token,
                                               float* __restrict__ out,
                                               float* __restrict__ o_cluster,
                                               float* __restrict__ o_embed,
                                               float* __restrict__ counts,
                                               float* __restrict__ scalars) {
    int n    = (blockIdx.x * 256 + threadIdx.x) >> 5;   // one wave per token
    int lane = threadIdx.x & 31;
    if (n >= NTOK) return;
    unsigned t = token[n];
    int b = n >> 10, m = n & 1023;
    const float* wrow = weight + (size_t)t * DDIM + lane * 8;
    float4 w0 = *(const float4*)(wrow);
    float4 w1 = *(const float4*)(wrow + 4);
    float wv[8] = { w0.x, w0.y, w0.z, w0.w, w1.x, w1.y, w1.z, w1.w };
    const float* zb = z   + (size_t)b * (DDIM * 1024) + m;
    float*       ob = out + (size_t)b * (DDIM * 1024) + m;
    float sq = 0.0f;
    #pragma unroll
    for (int j = 0; j < 8; ++j) {
        int c = lane * 8 + j;
        float zv = zb[(size_t)c * 1024];
        ob[(size_t)c * 1024] = wv[j];               // z_q (== STE output)
        float d = wv[j] - zv;
        sq += d * d;
        atomicAdd(&o_embed[(size_t)t * DDIM + c], ONEMD * zv);
    }
    #pragma unroll
    for (int mm = 16; mm >= 1; mm >>= 1) sq += __shfl_xor(sq, mm, 32);
    if (lane == 0) {
        atomicAdd(&scalars[0], sq);
        atomicAdd(&o_cluster[t], ONEMD);
        atomicAdd(&counts[t], 1.0f);
    }
}

// ---------------- kernel 6: codebook statistics ----------------
__global__ __launch_bounds__(256) void kstats(const float* __restrict__ counts,
                                              const float* __restrict__ o_cluster,
                                              float* __restrict__ scalars) {
    int k    = blockIdx.x * 256 + threadIdx.x;
    int lane = threadIdx.x & 31;
    float c  = counts[k];
    float used = (c > 0.0f) ? 1.0f : 0.0f;
    float p  = c * (1.0f / (float)NTOK);            // sum(counts) == NTOK exactly
    float ent = p * logf(p + 1e-10f);
    float cn  = o_cluster[k];
    #pragma unroll
    for (int m = 16; m >= 1; m >>= 1) {
        used += __shfl_xor(used, m, 32);
        ent  += __shfl_xor(ent,  m, 32);
        cn   += __shfl_xor(cn,   m, 32);
    }
    if (lane == 0) {
        atomicAdd(&scalars[1], used);
        atomicAdd(&scalars[2], ent);
        atomicAdd(&scalars[3], cn);                 // n = sum(cluster_new)
    }
}

// ---------------- kernel 7: weight_new + scalar outputs ----------------
__global__ __launch_bounds__(256) void finalize(const float* __restrict__ o_cluster,
                                                const float* __restrict__ o_embed,
                                                const float* __restrict__ scalars,
                                                float* __restrict__ o_weight,
                                                float* __restrict__ o_scal) {
    size_t i = (size_t)blockIdx.x * 256 + threadIdx.x;
    if (i < (size_t)KCODE * DDIM) {
        int k = (int)(i >> 8);
        float nsum = scalars[3];
        float sm = (o_cluster[k] + EPSF) / (nsum + (float)KCODE * EPSF) * nsum;
        o_weight[i] = o_embed[i] / sm;
    }
    if (i == 0) {
        float tsq = scalars[0];
        o_scal[0] = BETAF * tsq / (float)((size_t)NTOK * DDIM);  // commitment loss
        o_scal[1] = tsq / (float)NTOK;                           // quant_error
        o_scal[2] = scalars[1] / (float)KCODE;                   // utilization
        o_scal[3] = expf(-scalars[2]);                           // perplexity
    }
}

// ---------------- host launcher ----------------
extern "C" void kernel_launch(void* const* d_in, const int* in_sizes, int n_in,
                              void* d_out, int out_size, void* d_ws, size_t ws_size,
                              hipStream_t stream) {
    const float* z            = (const float*)d_in[0];   // (16,256,32,32)
    const float* weight       = (const float*)d_in[1];   // (16384,256)
    const float* cluster_size = (const float*)d_in[2];   // (16384,)
    const float* embed_avg    = (const float*)d_in[3];   // (16384,256)

    float* o         = (float*)d_out;
    float* o_out     = o;                                // 4194304
    float* o_scal    = o + 4194304;                      // loss, qerr, util, perpl
    float* o_weight  = o + 4194308;                      // 4194304
    float* o_cluster = o + 8388612;                      // 16384
    float* o_embed   = o + 8404996;                      // 4194304

    char* w = (char*)d_ws;
    unsigned short* zf_bf  = (unsigned short*)(w);                  // 8 MB
    unsigned short* w_bf   = (unsigned short*)(w + 8388608);        // 8 MB
    float*          wnorm  = (float*)(w + 16777216);                // 64 KB
    unsigned*       token  = (unsigned*)(w + 16842752);             // 64 KB
    float*          counts = (float*)(w + 16908288);                // 64 KB
    float*          scal   = (float*)(w + 16973824);                // 16 B

    initk<<<16384, 256, 0, stream>>>(cluster_size, embed_avg, o_cluster, o_embed, counts, scal);
    wprep<<<2048, 256, 0, stream>>>(weight, w_bf, wnorm);
    ztrans<<<dim3(32, 8, 16), dim3(32, 8), 0, stream>>>(z, zf_bf);
    gemm_argmin<<<NTOK / MTILE, GBLK, 0, stream>>>(zf_bf, w_bf, wnorm, token);
    scatter<<<2048, 256, 0, stream>>>(z, weight, token, o_out, o_cluster, o_embed, counts, scal);
    kstats<<<KCODE / 256, 256, 0, stream>>>(counts, o_cluster, scal);
    finalize<<<16384, 256, 0, stream>>>(o_cluster, o_embed, scal, o_weight, o_scal);
}
